// BigNet_42288247996850
// MI455X (gfx1250) — matrix operand order repaired and checked by reading, not compiled, not run on hardware
//
#include <hip/hip_runtime.h>

// ---------------------------------------------------------------------------
// Types for CDNA5 WMMA (wave32, bf16 in / f32 out, 16x16x32)
// ---------------------------------------------------------------------------
typedef __bf16 bf16_t;
typedef __attribute__((ext_vector_type(16))) __bf16 v16bf;
typedef __attribute__((ext_vector_type(8)))  __bf16 v8bf;
typedef __attribute__((ext_vector_type(8)))  float  v8f;

#define D_DIM   256
#define N_NODES 100000
#define N_EDGES 320000
#define N_GRAPHS 64
#define OUT_DIM 978
#define KZ      544   // 514 padded to 17*32
#define ZSTR    552   // LDS row stride (bf16 elems), 16B aligned
#define HSTR    264   // LDS row stride for 256-wide tiles
#define ETILE   64    // edges per CGConv block  (320000/64 = 5000)
#define GTILE   32    // nodes per GAT GEMM block (100000/32 = 3125)

#define WMMA_BF16(a, b, c) \
  __builtin_amdgcn_wmma_f32_16x16x32_bf16(false, (a), false, (b), (short)0, (c), false, false)

// ---------------------------------------------------------------------------
// A-fragment load from an LDS row-major bf16 tile.
// 16-bit A 16x32 layout (ISA 7.12.2): lanes 0-15 -> M=0..15, K low half;
// lanes 16-31 -> same M, K high half. VGPRs 0..3 hold K=half*8+0..7 (pairs),
// VGPRs 4..7 hold K=16+half*8+0..7.  -> two 16-byte LDS loads per fragment.
// ---------------------------------------------------------------------------
__device__ inline v16bf load_a_frag(const bf16_t* zbase, int strideElems, int kbase) {
  int lane = threadIdx.x & 31;
  int m    = lane & 15;
  int half = lane >> 4;
  const bf16_t* row = zbase + m * strideElems + kbase + half * 8;
  v8bf lo = *(const v8bf*)(row);        // k = kbase + half*8 + 0..7
  v8bf hi = *(const v8bf*)(row + 16);   // k = kbase + 16 + half*8 + 0..7
  v16bf a;
#pragma unroll
  for (int i = 0; i < 8; ++i) { a[i] = lo[i]; a[8 + i] = hi[i]; }
  return a;
}

__device__ inline void atomicMaxFloat(float* addr, float value) {
  if (value >= 0.0f) atomicMax((int*)addr, __float_as_int(value));
  else               atomicMin((unsigned int*)addr, __float_as_uint(value));
}

// ---------------------------------------------------------------------------
// Weight pre-pack: f32 [K, 256] row-major -> bf16 B-fragment-major:
// out[(((kc*16 + nt)*32 + lane)*16 + e] = W[kc*32 + (e<8 ? (lane>>4)*8 + e : 16 + (lane>>4)*8 + e-8)][nt*16 + (lane&15)]
// (zero-padded past K). Each WMMA B fragment is then one 32B contiguous load/lane.
// ---------------------------------------------------------------------------
__global__ void pack_w_kernel(const float* __restrict__ W, bf16_t* __restrict__ Wp,
                              int K, int Kc) {
  long i = (long)blockIdx.x * blockDim.x + threadIdx.x;
  long total = (long)Kc * 16 * 32 * 16;
  if (i >= total) return;
  int e    = (int)(i & 15);
  int lane = (int)((i >> 4) & 31);
  long rest = i >> 9;
  int nt = (int)(rest & 15);
  int kc = (int)(rest >> 4);
  int k  = kc * 32 + ((e < 8) ? (lane >> 4) * 8 + e : 16 + (lane >> 4) * 8 + (e - 8));
  int n  = nt * 16 + (lane & 15);
  float v = (k < K) ? W[(long)k * D_DIM + n] : 0.0f;
  Wp[i] = (bf16_t)v;
}

// ---------------------------------------------------------------------------
// Node init: x = emb[gidx] + acts @ pe_W + pe_b ; also bf16 copy and CG accum.
// ---------------------------------------------------------------------------
__global__ void node_init_kernel(const int* __restrict__ gidx,
                                 const float* __restrict__ acts,
                                 const float* __restrict__ emb,
                                 const float* __restrict__ peW,
                                 const float* __restrict__ peb,
                                 float* __restrict__ x, bf16_t* __restrict__ xb,
                                 float* __restrict__ hcg) {
  long i = (long)blockIdx.x * blockDim.x + threadIdx.x;
  if (i >= (long)N_NODES * D_DIM) return;
  long node = i >> 8; int d = (int)(i & 255);
  float a0 = acts[node * 2], a1 = acts[node * 2 + 1];
  float v = emb[(long)gidx[node] * D_DIM + d] + a0 * peW[d] + a1 * peW[D_DIM + d] + peb[d];
  x[i] = v; xb[i] = (bf16_t)v; hcg[i] = v;
}

__global__ void fill_kernel(float* __restrict__ p, float v, long n) {
  long i = (long)blockIdx.x * blockDim.x + threadIdx.x;
  if (i < n) p[i] = v;
}

// ---------------------------------------------------------------------------
// CGConv: 64-edge tile/block. z = [x_dst | x_src | sign | 0pad] in LDS
// (64 x 544 bf16). Each wave owns two 16-wide N tiles; per K-chunk one pair
// of B fragments (Wf, Ws) feeds 4 M-tiles -> 8 WMMAs, amortizing weight
// traffic 4x vs a 16-edge tile. msg = sigmoid(f)*softplus(s) in-register,
// then f32 atomic scatter-add by dst.
// ---------------------------------------------------------------------------
__global__ __launch_bounds__(256)
void cgconv_kernel(const bf16_t* __restrict__ xb,
                   const int* __restrict__ src, const int* __restrict__ dst,
                   const float* __restrict__ sign,
                   const bf16_t* __restrict__ WfP, const bf16_t* __restrict__ WsP,
                   const float* __restrict__ bf_, const float* __restrict__ bs_,
                   float* __restrict__ hout) {
  __shared__ bf16_t z[ETILE * ZSTR];           // 70656 B
  __shared__ int dloc[ETILE], sloc[ETILE];
  int t  = threadIdx.x;
  long e0 = (long)blockIdx.x * ETILE;
  if (t < ETILE) { dloc[t] = dst[e0 + t]; sloc[t] = src[e0 + t]; }
  __syncthreads();
  // vectorized fill: 64 rows x 68 chunks of 8 bf16 (16B each)
  for (int idx = t; idx < ETILE * 68; idx += 256) {
    int e = idx / 68, c = idx - e * 68;
    int f = c * 8;
    v8bf v;
    if (f < D_DIM) {
      v = *(const v8bf*)&xb[(long)dloc[e] * D_DIM + f];               // x_i = x[dst]
    } else if (f < 2 * D_DIM) {
      v = *(const v8bf*)&xb[(long)sloc[e] * D_DIM + (f - D_DIM)];     // x_j = x[src]
    } else {
#pragma unroll
      for (int q = 0; q < 8; ++q) v[q] = (bf16_t)0.0f;
      if (f == 2 * D_DIM) {
        v[0] = (bf16_t)sign[(e0 + e) * 2];
        v[1] = (bf16_t)sign[(e0 + e) * 2 + 1];
      }
    }
    *(v8bf*)&z[e * ZSTR + f] = v;
  }
  __syncthreads();

  int wave = t >> 5, lane = t & 31;
  int ncol = lane & 15, half = lane >> 4;
#pragma unroll 1
  for (int pass = 0; pass < 2; ++pass) {
    int nt = wave + pass * 8;           // 0..15 -> output cols nt*16..nt*16+15
    v8f af[4], as_[4];
#pragma unroll
    for (int mt = 0; mt < 4; ++mt) { af[mt] = (v8f){}; as_[mt] = (v8f){}; }
#pragma unroll 1
    for (int kc = 0; kc < 17; ++kc) {
      v16bf bF = *(const v16bf*)(WfP + (((long)kc * 16 + nt) * 32 + lane) * 16);
      v16bf bS = *(const v16bf*)(WsP + (((long)kc * 16 + nt) * 32 + lane) * 16);
#pragma unroll
      for (int mt = 0; mt < 4; ++mt) {
        v16bf a = load_a_frag(z + mt * 16 * ZSTR, ZSTR, kc * 32);
        af[mt]  = WMMA_BF16(a, bF, af[mt]);
        as_[mt] = WMMA_BF16(a, bS, as_[mt]);
      }
    }
    int n = nt * 16 + ncol;
    float bfv = bf_[n], bsv = bs_[n];
#pragma unroll
    for (int mt = 0; mt < 4; ++mt) {
#pragma unroll
      for (int r = 0; r < 8; ++r) {
        int m = mt * 16 + r + half * 8;          // edge row within tile
        float f = af[mt][r] + bfv;
        float s = as_[mt][r] + bsv;
        float sig = __builtin_amdgcn_rcpf(1.0f + __expf(-f));   // v_rcp_f32
        float sp  = (s > 20.0f) ? s : __logf(1.0f + __expf(s)); // softplus
        atomicAdd(&hout[(long)dloc[m] * D_DIM + n], sig * sp);
      }
    }
  }
}

// ---------------------------------------------------------------------------
// GAT GEMM: 32-node tile/block, hW = hb @ W (K=256, 8 WMMA chunks), one B
// fragment feeds 2 M-tiles; direct stores (no atomics).
// ---------------------------------------------------------------------------
__global__ __launch_bounds__(256)
void gat_gemm_kernel(const bf16_t* __restrict__ hb, const bf16_t* __restrict__ WgP,
                     float* __restrict__ hW) {
  __shared__ bf16_t tile[GTILE * HSTR];
  int t = threadIdx.x;
  long n0 = (long)blockIdx.x * GTILE;
  for (int idx = t; idx < GTILE * 32; idx += 256) {   // 32 chunks of 8 bf16/row
    int r = idx >> 5, c = (idx & 31) * 8;
    *(v8bf*)&tile[r * HSTR + c] = *(const v8bf*)&hb[(n0 + r) * D_DIM + c];
  }
  __syncthreads();
  int wave = t >> 5, lane = t & 31;
  int half = lane >> 4;
#pragma unroll 1
  for (int pass = 0; pass < 2; ++pass) {
    int nt = wave + pass * 8;
    v8f acc0 = {}, acc1 = {};
#pragma unroll
    for (int kc = 0; kc < 8; ++kc) {
      v16bf b  = *(const v16bf*)(WgP + (((long)kc * 16 + nt) * 32 + lane) * 16);
      v16bf a0 = load_a_frag(tile, HSTR, kc * 32);
      v16bf a1 = load_a_frag(tile + 16 * HSTR, HSTR, kc * 32);
      acc0 = WMMA_BF16(a0, b, acc0);
      acc1 = WMMA_BF16(a1, b, acc1);
    }
    int n = nt * 16 + (lane & 15);
#pragma unroll
    for (int r = 0; r < 8; ++r) {
      int m = r + half * 8;
      hW[(n0 + m) * D_DIM + n]      = acc0[r];
      hW[(n0 + 16 + m) * D_DIM + n] = acc1[r];
    }
  }
}

// Per-node attention dots: asrc[n] = hW[n,:]·a_src ; adst[n] = hW[n,:]·a_dst
__global__ __launch_bounds__(256)
void att_dots_kernel(const float* __restrict__ hW,
                     const float* __restrict__ asv, const float* __restrict__ adv,
                     float* __restrict__ asrc, float* __restrict__ adst) {
  long node = (long)blockIdx.x * 8 + (threadIdx.x >> 5);
  int lane = threadIdx.x & 31;
  if (node >= N_NODES) return;
  float s1 = 0.0f, s2 = 0.0f;
  for (int d = lane; d < D_DIM; d += 32) {
    float h = hW[node * D_DIM + d];
    s1 += h * asv[d]; s2 += h * adv[d];
  }
#pragma unroll
  for (int off = 16; off > 0; off >>= 1) {
    s1 += __shfl_down(s1, off, 32);
    s2 += __shfl_down(s2, off, 32);
  }
  if (lane == 0) { asrc[node] = s1; adst[node] = s2; }
}

// Edge logits (self loops appended) + segment max by dst
__global__ void logit_kernel(const int* __restrict__ src, const int* __restrict__ dst,
                             const float* __restrict__ asrc, const float* __restrict__ adst,
                             float* __restrict__ logit, float* __restrict__ mmax) {
  int i = blockIdx.x * blockDim.x + threadIdx.x;
  const int Esl = N_EDGES + N_NODES;
  if (i >= Esl) return;
  int s = (i < N_EDGES) ? src[i] : (i - N_EDGES);
  int d = (i < N_EDGES) ? dst[i] : (i - N_EDGES);
  float l = asrc[s] + adst[d];
  l = (l >= 0.0f) ? l : 0.2f * l;          // leaky_relu(0.2)
  logit[i] = l;
  atomicMaxFloat(&mmax[d], l);
}

// ex = exp(logit - max[dst]) (in place), denom[dst] += ex
__global__ void expsum_kernel(const int* __restrict__ dst,
                              float* __restrict__ elog, const float* __restrict__ mmax,
                              float* __restrict__ denom) {
  int i = blockIdx.x * blockDim.x + threadIdx.x;
  const int Esl = N_EDGES + N_NODES;
  if (i >= Esl) return;
  int d = (i < N_EDGES) ? dst[i] : (i - N_EDGES);
  float ex = __expf(elog[i] - mmax[d]);
  elog[i] = ex;
  atomicAdd(&denom[d], ex);
}

// g[dst,:] += hW[src,:] * (ex/denom[dst]) -- one block per edge, one lane per col
__global__ __launch_bounds__(256)
void gat_scatter_kernel(const int* __restrict__ src, const int* __restrict__ dst,
                        const float* __restrict__ elog, const float* __restrict__ denom,
                        const float* __restrict__ hW, float* __restrict__ g) {
  int e = blockIdx.x;
  int c = threadIdx.x;
  int s = (e < N_EDGES) ? src[e] : (e - N_EDGES);
  int d = (e < N_EDGES) ? dst[e] : (e - N_EDGES);
  float coef = elog[e] * __builtin_amdgcn_rcpf(denom[d]);
  atomicAdd(&g[(long)d * D_DIM + c], hW[(long)s * D_DIM + c] * coef);
}

// h_out = prelu( (gin + bias)/sqrt(1+eps)*gamma + beta + resid, alpha ) ; + bf16 copy
__global__ void epilogue_kernel(const float* __restrict__ gin,
                                const float* __restrict__ bias,     // may be null
                                const float* __restrict__ gamma,
                                const float* __restrict__ beta,
                                const float* __restrict__ alpha_p,  // scalar on device
                                const float* __restrict__ resid,
                                float* __restrict__ hout, bf16_t* __restrict__ hbout) {
  long i = (long)blockIdx.x * blockDim.x + threadIdx.x;
  if (i >= (long)N_NODES * D_DIM) return;
  int d = (int)(i & 255);
  float a = *alpha_p;
  const float inv = rsqrtf(1.0f + 1e-5f);
  float v = gin[i] + (bias ? bias[d] : 0.0f);
  v = v * inv * gamma[d] + beta[d] + resid[i];
  v = (v >= 0.0f) ? v : a * v;
  hout[i] = v; hbout[i] = (bf16_t)v;
}

// Pool input features x by graph id
__global__ void pool_kernel(const float* __restrict__ x, const int* __restrict__ batch,
                            float* __restrict__ pooled) {
  long i = (long)blockIdx.x * blockDim.x + threadIdx.x;
  if (i >= (long)N_NODES * D_DIM) return;
  long node = i >> 8; int d = (int)(i & 255);
  atomicAdd(&pooled[(long)batch[node] * D_DIM + d], x[i]);
}

__global__ void fc1_kernel(const float* __restrict__ pooled, const float* __restrict__ W,
                           const float* __restrict__ b, float* __restrict__ z1) {
  int i = blockIdx.x * blockDim.x + threadIdx.x;
  if (i >= N_GRAPHS * 512) return;
  int g = i >> 9, j = i & 511;
  float s = b[j];
  for (int k = 0; k < D_DIM; ++k) s += pooled[g * D_DIM + k] * W[k * 512 + j];
  z1[i] = fmaxf(s, 0.0f);
}

__global__ void fc2_kernel(const float* __restrict__ z1, const float* __restrict__ W,
                           const float* __restrict__ b, float* __restrict__ logits) {
  int i = blockIdx.x * blockDim.x + threadIdx.x;
  if (i >= N_GRAPHS * OUT_DIM) return;
  int g = i / OUT_DIM, j = i - g * OUT_DIM;
  float s = b[j];
  for (int k = 0; k < 512; ++k) s += z1[g * 512 + k] * W[k * OUT_DIM + j];
  logits[i] = s;
}

__global__ __launch_bounds__(256)
void logsoftmax_kernel(const float* __restrict__ logits, float* __restrict__ out) {
  __shared__ float red[256];
  int g = blockIdx.x, t = threadIdx.x;
  float m = -INFINITY;
  for (int j = t; j < OUT_DIM; j += 256) m = fmaxf(m, logits[g * OUT_DIM + j]);
  red[t] = m; __syncthreads();
  for (int off = 128; off > 0; off >>= 1) {
    if (t < off) red[t] = fmaxf(red[t], red[t + off]);
    __syncthreads();
  }
  m = red[0]; __syncthreads();
  float s = 0.0f;
  for (int j = t; j < OUT_DIM; j += 256) s += __expf(logits[g * OUT_DIM + j] - m);
  red[t] = s; __syncthreads();
  for (int off = 128; off > 0; off >>= 1) {
    if (t < off) red[t] += red[t + off];
    __syncthreads();
  }
  float lse = m + __logf(red[0]);
  for (int j = t; j < OUT_DIM; j += 256) out[g * OUT_DIM + j] = logits[g * OUT_DIM + j] - lse;
}

// ---------------------------------------------------------------------------
// Host launcher
// ---------------------------------------------------------------------------
extern "C" void kernel_launch(void* const* d_in, const int* in_sizes, int n_in,
                              void* d_out, int out_size, void* d_ws, size_t ws_size,
                              hipStream_t stream) {
  const int*   gidx   = (const int*)  d_in[0];
  const float* acts   = (const float*)d_in[1];
  const int*   eidx   = (const int*)  d_in[2];
  const float* sign   = (const float*)d_in[3];
  const int*   batch  = (const int*)  d_in[4];
  const float* emb    = (const float*)d_in[5];
  const float* peW    = (const float*)d_in[6];
  const float* peb    = (const float*)d_in[7];
  const float* cgWf   = (const float*)d_in[8];
  const float* cgbf   = (const float*)d_in[9];
  const float* cgWs   = (const float*)d_in[10];
  const float* cgbs   = (const float*)d_in[11];
  const float* gatW   = (const float*)d_in[12];
  const float* gatAs  = (const float*)d_in[13];
  const float* gatAd  = (const float*)d_in[14];
  const float* gatB   = (const float*)d_in[15];
  const float* bnG    = (const float*)d_in[16];
  const float* bnB    = (const float*)d_in[17];
  const float* alpha  = (const float*)d_in[18];
  const float* fc1W   = (const float*)d_in[19];
  const float* fc1b   = (const float*)d_in[20];
  const float* fc2W   = (const float*)d_in[21];
  const float* fc2b   = (const float*)d_in[22];

  const int* srcp = eidx;
  const int* dstp = eidx + N_EDGES;
  const long ND = (long)N_NODES * D_DIM;
  const int  Esl = N_EDGES + N_NODES;

  char* base = (char*)d_ws;
  size_t off = 0;
  auto carve = [&](size_t bytes) -> void* {
    void* p = base + off;
    off = (off + bytes + 255) & ~(size_t)255;
    return p;
  };
  float*  x    = (float*) carve(ND * 4);
  bf16_t* xb   = (bf16_t*)carve(ND * 2);        // reused as hb after CGConv
  float*  hA   = (float*) carve(ND * 4);
  float*  hB   = (float*) carve(ND * 4);
  float*  hW   = (float*) carve(ND * 4);
  float*  g    = (float*) carve(ND * 4);
  bf16_t* WfP  = (bf16_t*)carve((size_t)17 * 16 * 32 * 16 * 2);
  bf16_t* WsP  = (bf16_t*)carve((size_t)17 * 16 * 32 * 16 * 2);
  bf16_t* WgP  = (bf16_t*)carve((size_t)4 * 8 * 16 * 32 * 16 * 2);
  float*  asrc = (float*) carve((size_t)N_NODES * 4);
  float*  adst = (float*) carve((size_t)N_NODES * 4);
  float*  mmax = (float*) carve((size_t)N_NODES * 4);
  float*  den  = (float*) carve((size_t)N_NODES * 4);
  float*  elog = (float*) carve((size_t)Esl * 4);
  float*  pooled = (float*)carve((size_t)N_GRAPHS * D_DIM * 4);
  float*  z1     = (float*)carve((size_t)N_GRAPHS * 512 * 4);
  float*  lgt    = (float*)carve((size_t)N_GRAPHS * OUT_DIM * 4);
  bf16_t* hb = xb;

  // --- weight pre-pack (bf16, fragment-major) ---
  {
    long tcg = (long)17 * 16 * 32 * 16;
    pack_w_kernel<<<(int)((tcg + 255) / 256), 256, 0, stream>>>(cgWf, WfP, 514, 17);
    pack_w_kernel<<<(int)((tcg + 255) / 256), 256, 0, stream>>>(cgWs, WsP, 514, 17);
    long tg = (long)8 * 16 * 32 * 16;      // per layer
    for (int l = 0; l < 4; ++l)
      pack_w_kernel<<<(int)((tg + 255) / 256), 256, 0, stream>>>(
          gatW + (long)l * D_DIM * D_DIM, WgP + (long)l * tg, 256, 8);
  }

  // --- node features: x, xb, CG accumulator (h = x + sum msg) ---
  node_init_kernel<<<(int)(ND / 256), 256, 0, stream>>>(gidx, acts, emb, peW, peb, x, xb, hA);

  // --- CGConv (WMMA) + scatter into hA ---
  cgconv_kernel<<<N_EDGES / ETILE, 256, 0, stream>>>(xb, srcp, dstp, sign, WfP, WsP, cgbf, cgbs, hA);

  // --- h = prelu(bn(hA) + x) -> hB, hb ---
  epilogue_kernel<<<(int)(ND / 256), 256, 0, stream>>>(hA, nullptr, bnG, bnB, alpha, x, hB, hb);

  float* hcur = hB;
  float* hnxt = hA;
  for (int l = 0; l < 4; ++l) {
    // hW = h @ W_l  (WMMA)
    gat_gemm_kernel<<<N_NODES / GTILE, 256, 0, stream>>>(hb, WgP + (long)l * 8 * 16 * 32 * 16, hW);
    // attention dot products
    att_dots_kernel<<<(N_NODES + 7) / 8, 256, 0, stream>>>(
        hW, gatAs + l * D_DIM, gatAd + l * D_DIM, asrc, adst);
    // init segment buffers
    fill_kernel<<<(N_NODES + 255) / 256, 256, 0, stream>>>(mmax, -INFINITY, N_NODES);
    fill_kernel<<<(N_NODES + 255) / 256, 256, 0, stream>>>(den, 0.0f, N_NODES);
    fill_kernel<<<(int)(ND / 256), 256, 0, stream>>>(g, 0.0f, ND);
    // softmax over incoming edges
    logit_kernel<<<(Esl + 255) / 256, 256, 0, stream>>>(srcp, dstp, asrc, adst, elog, mmax);
    expsum_kernel<<<(Esl + 255) / 256, 256, 0, stream>>>(dstp, elog, mmax, den);
    // weighted aggregation
    gat_scatter_kernel<<<Esl, 256, 0, stream>>>(srcp, dstp, elog, den, hW, g);
    // h = prelu(bn(g + bias) + h) -> hnxt, hb
    epilogue_kernel<<<(int)(ND / 256), 256, 0, stream>>>(
        g, gatB + l * D_DIM, bnG + (l + 1) * D_DIM, bnB + (l + 1) * D_DIM,
        alpha + (l + 1), hcur, hnxt, hb);
    float* tmp = hcur; hcur = hnxt; hnxt = tmp;
  }

  // --- readout head (pools ORIGINAL x, matching the reference) ---
  fill_kernel<<<(N_GRAPHS * D_DIM + 255) / 256, 256, 0, stream>>>(pooled, 0.0f, (long)N_GRAPHS * D_DIM);
  pool_kernel<<<(int)(ND / 256), 256, 0, stream>>>(x, batch, pooled);
  fc1_kernel<<<(N_GRAPHS * 512 + 255) / 256, 256, 0, stream>>>(pooled, fc1W, fc1b, z1);
  fc2_kernel<<<(N_GRAPHS * OUT_DIM + 255) / 256, 256, 0, stream>>>(z1, fc2W, fc2b, lgt);
  logsoftmax_kernel<<<N_GRAPHS, 256, 0, stream>>>(lgt, (float*)d_out);
}
